// CustomCharBiLSTM_37323265802559
// MI455X (gfx1250) — matrix operand order repaired and checked by reading, not compiled, not run on hardware
//
#include <hip/hip_runtime.h>
#include <hip/hip_bf16.h>

typedef __attribute__((ext_vector_type(16))) _Float16 v16h;
typedef __attribute__((ext_vector_type(8)))  float    v8f;

#define N_WORDS 16384
#define MAX_LEN 16
#define VOCAB   128
#define EMB     64
#define HID     256
#define WPB     32     // words per block (2 M-tiles of 16)
#define MT      2      // M tiles per block
#define KT_STRIDE (2 * 64 * 512)   // f16 elems per k-slice (both dirs, 64 nt)

__device__ __forceinline__ v8f zf8() { v8f z = {}; return z; }
// sigmoid / tanh via single v_exp_f32 each
__device__ __forceinline__ float fsigmoid(float x) { return 1.f / (1.f + __expf(-x)); }
__device__ __forceinline__ float ftanh(float x)    { return 2.f / (1.f + __expf(-2.f * x)) - 1.f; }

// ---------------------------------------------------------------------------
// Weight prep: Wcat[K=320][N=1024] per direction, K<64 from W_ih (1024x64
// row-major), K>=64 from W_hh (1024x256 row-major), f16, in WMMA B-fragment
// order. Layout is KT-MAJOR so the main loop gets one base pointer per
// k-slice and constant-immediate offsets for the 8 gate tiles:
//   wbuf[kt][dir][nt][lane*16+j], lane = half*16+col, element j (0..15):
//   K = kt*32 + (j<8 ? half*8+j : 16+half*8+(j-8)), N = nt*16 + col.
// ---------------------------------------------------------------------------
__global__ void prep_weights(const float* __restrict__ Wih_f,
                             const float* __restrict__ Whh_f,
                             const float* __restrict__ Wih_b,
                             const float* __restrict__ Whh_b,
                             _Float16* __restrict__ wbuf) {
  int bid = blockIdx.x;            // 2 * 64 * 10 blocks
  int dir = bid / 640;
  int rem = bid % 640;
  int nt = rem / 10, kt = rem % 10;
  int lane = threadIdx.x;
  int half = lane >> 4, col = lane & 15;
  int N = nt * 16 + col;
  const float* Wih = dir ? Wih_b : Wih_f;
  const float* Whh = dir ? Whh_b : Whh_f;
  _Float16* p = wbuf + (((kt * 2 + dir) * 64 + nt) * 512 + lane * 16);
#pragma unroll
  for (int j = 0; j < 16; ++j) {
    int K = kt * 32 + ((j < 8) ? (half * 8 + j) : (16 + half * 8 + (j - 8)));
    float w = (K < EMB) ? Wih[N * EMB + K] : Whh[N * HID + (K - EMB)];
    p[j] = (_Float16)w;
  }
}

__global__ void prep_bias(const float* __restrict__ bih_f,
                          const float* __restrict__ bhh_f,
                          const float* __restrict__ bih_b,
                          const float* __restrict__ bhh_b,
                          float* __restrict__ bcat) {
  int i = blockIdx.x * blockDim.x + threadIdx.x;  // 2048 total
  if (i < 1024) {
    bcat[i] = bih_f[i] + bhh_f[i];
  } else if (i < 2048) {
    int j = i - 1024;
    bcat[i] = bih_b[j] + bhh_b[j];
  }
}

// ---------------------------------------------------------------------------
// Main BiLSTM kernel. One block = 32 words (2 M-tiles), 512 threads = 16
// waves. Waves 0-7: forward; 8-15: backward. Wave w8 owns hidden units
// [w8*32, w8*32+32) => gate tiles nt = g*16 + w8*2 + t, and both M-tiles, so
// each B fragment feeds two WMMAs. Cell state c lives in registers; h lives
// in LDS (f16, A-fragment layout) -- masked carry = "store only if mask".
// Weight loads use a per-step K-slice rotation so they are NOT loop-
// invariant (prevents hoist+spill); per k-slice one base pointer + constant
// IOFFSETs for the 8 fragments.
// ---------------------------------------------------------------------------
__launch_bounds__(512, 1)
__global__ void bilstm_kernel(const int* __restrict__ char_ids,
                              const int* __restrict__ lengths,
                              const float* __restrict__ emb,
                              const _Float16* __restrict__ wbuf,
                              const float* __restrict__ bcat,
                              float* __restrict__ out) {
  __shared__ __align__(32) _Float16 emb_lds[VOCAB * EMB];          // 16 KB
  __shared__ __align__(32) _Float16 zbuf[2][MT][10][32][16];       // 40 KB

  const int tid  = threadIdx.x;
  const int wave = tid >> 5, lane = tid & 31;
  const int dir  = wave >> 3, w8 = wave & 7;
  const int half = lane >> 4, col = lane & 15;
  const int wb   = blockIdx.x * WPB;

  // Embedding table -> LDS (f16); zero z buffer (h starts at 0).
  for (int i = tid; i < VOCAB * EMB; i += 512) emb_lds[i] = (_Float16)emb[i];
  for (int i = tid; i < 2 * MT * 10 * 32 * 16; i += 512)
    ((_Float16*)zbuf)[i] = (_Float16)0.f;

  // Persistent cell state: cR[m][t*8+r] -> word wb+m*16+r+8*half,
  // hidden w8*32 + t*16 + col (matches f32 C/D fragment layout).
  float cR[MT][16];
#pragma unroll
  for (int m = 0; m < MT; ++m)
#pragma unroll
    for (int i = 0; i < 16; ++i) cR[m][i] = 0.f;

  int lenR[MT][8];
#pragma unroll
  for (int m = 0; m < MT; ++m)
#pragma unroll
    for (int r = 0; r < 8; ++r) lenR[m][r] = lengths[wb + m * 16 + r + 8 * half];

  float biasR[4][2];
#pragma unroll
  for (int g = 0; g < 4; ++g)
#pragma unroll
    for (int t = 0; t < 2; ++t)
      biasR[g][t] = bcat[dir * 1024 + g * 256 + w8 * 32 + t * 16 + col];

  // Per-wave/lane weight base; per k-slice add kt*KT_STRIDE, then the 8
  // gate-tile fragments are at constant offsets (g*16+t)*512 f16.
  const _Float16* wlane =
      wbuf + (size_t)((dir * 64 + w8 * 2) * 512 + lane * 16);
  __builtin_prefetch(wlane, 0, 1);   // global_prefetch_b8 (L2 warm)

  // x-writer waves: w8 < MT; wave (dir, w8=m) materializes x for M-tile m.
  const int mx = w8;
  int len_x = 0;
  if (w8 < MT) len_x = lengths[wb + mx * 16 + col];

  auto write_x = [&](int s) {
    int word = wb + mx * 16 + col;                  // A row M = col
    int ts = (dir == 0) ? s : (len_x - 1 - s);
    if (ts < 0) ts = 0;
    int cid = char_ids[word * MAX_LEN + ts];
#pragma unroll
    for (int kt = 0; kt < 2; ++kt)
#pragma unroll
      for (int j = 0; j < 16; ++j) {
        int K = kt * 32 + ((j < 8) ? (half * 8 + j) : (16 + half * 8 + (j - 8)));
        zbuf[dir][mx][kt][lane][j] = emb_lds[cid * EMB + K];
      }
  };

  __syncthreads();
  if (w8 < MT) write_x(0);
  __syncthreads();

#pragma unroll 1
  for (int s = 0; s < MAX_LEN; ++s) {
    const int srot = s % 10;   // K-slice rotation: weight addrs depend on s

    v8f acc[4][2][MT];
#pragma unroll
    for (int g = 0; g < 4; ++g)
#pragma unroll
      for (int t = 0; t < 2; ++t)
#pragma unroll
        for (int m = 0; m < MT; ++m) acc[g][t][m] = zf8();

    // gates(32x1024 slice) = z(32x320) @ Wcat(320x1024 slice)
#pragma unroll
    for (int kk = 0; kk < 10; ++kk) {
      int kt = kk + srot;
      if (kt >= 10) kt -= 10;
      const _Float16* bp = wlane + (size_t)kt * KT_STRIDE;
      v16h a0 = *(const v16h*)(&zbuf[dir][0][kt][lane][0]);
      v16h a1 = *(const v16h*)(&zbuf[dir][1][kt][lane][0]);
#pragma unroll
      for (int g = 0; g < 4; ++g) {
#pragma unroll
        for (int t = 0; t < 2; ++t) {
          v16h b = *(const v16h*)(bp + (g * 16 + t) * 512);  // const IOFFSET
          acc[g][t][0] = __builtin_amdgcn_wmma_f32_16x16x32_f16(
              false, a0, false, b, (short)0, acc[g][t][0], false, false);
          acc[g][t][1] = __builtin_amdgcn_wmma_f32_16x16x32_f16(
              false, a1, false, b, (short)0, acc[g][t][1], false, false);
        }
      }
    }

    __syncthreads();  // all zbuf reads for this step done

    // LSTM cell update; h goes straight to LDS (store-if-mask = carry).
#pragma unroll
    for (int m = 0; m < MT; ++m)
#pragma unroll
      for (int t = 0; t < 2; ++t)
#pragma unroll
        for (int r = 0; r < 8; ++r) {
          float gi = acc[0][t][m][r] + biasR[0][t];
          float gf = acc[1][t][m][r] + biasR[1][t];
          float gg = acc[2][t][m][r] + biasR[2][t];
          float go = acc[3][t][m][r] + biasR[3][t];
          float c2 = fsigmoid(gf) * cR[m][t * 8 + r] + fsigmoid(gi) * ftanh(gg);
          float h2 = fsigmoid(go) * ftanh(c2);
          if (s < lenR[m][r]) {
            cR[m][t * 8 + r] = c2;
            // h -> A-fragment slot: K = 64 + w8*32 + t*16 + col -> kt = 2+w8
            int hz = col >> 3;
            int j  = (t ? 8 : 0) + (col & 7);
            int lz = hz * 16 + r + 8 * half;
            zbuf[dir][m][2 + w8][lz][j] = (_Float16)h2;
          }
        }

    if (w8 < MT && s + 1 < MAX_LEN) write_x(s + 1);
    __syncthreads();
  }

  // Output: read final h back from LDS. out[word][0..255]=h_f, [256..511]=h_b
#pragma unroll
  for (int m = 0; m < MT; ++m)
#pragma unroll
    for (int t = 0; t < 2; ++t)
#pragma unroll
      for (int r = 0; r < 8; ++r) {
        int hz = col >> 3;
        int j  = (t ? 8 : 0) + (col & 7);
        int lz = hz * 16 + r + 8 * half;
        float hv = (float)zbuf[dir][m][2 + w8][lz][j];
        int word = wb + m * 16 + r + 8 * half;
        int hid  = w8 * 32 + t * 16 + col;
        out[word * 512 + dir * 256 + hid] = hv;
      }
}

extern "C" void kernel_launch(void* const* d_in, const int* in_sizes, int n_in,
                              void* d_out, int out_size, void* d_ws, size_t ws_size,
                              hipStream_t stream) {
  const int*   char_ids = (const int*)d_in[0];
  const int*   lengths  = (const int*)d_in[1];
  const float* emb      = (const float*)d_in[2];
  const float* Wih_f    = (const float*)d_in[3];
  const float* Whh_f    = (const float*)d_in[4];
  const float* bih_f    = (const float*)d_in[5];
  const float* bhh_f    = (const float*)d_in[6];
  const float* Wih_b    = (const float*)d_in[7];
  const float* Whh_b    = (const float*)d_in[8];
  const float* bih_b    = (const float*)d_in[9];
  const float* bhh_b    = (const float*)d_in[10];
  float* out = (float*)d_out;

  // Workspace layout: [wbuf: 10*2*64*512 f16 = 1.25 MiB][bcat: 2048 f32]
  _Float16* wbuf = (_Float16*)d_ws;
  float* bcat = (float*)((char*)d_ws + (size_t)10 * KT_STRIDE * sizeof(_Float16));

  prep_weights<<<1280, 32, 0, stream>>>(Wih_f, Whh_f, Wih_b, Whh_b, wbuf);
  prep_bias<<<8, 256, 0, stream>>>(bih_f, bhh_f, bih_b, bhh_b, bcat);
  bilstm_kernel<<<N_WORDS / WPB, 512, 0, stream>>>(char_ids, lengths, emb,
                                                   wbuf, bcat, out);
}
